// SoftDecisionTree_11862699672110
// MI455X (gfx1250) — matrix operand order repaired and checked by reading, not compiled, not run on hardware
//
#include <hip/hip_runtime.h>
#include <hip/hip_bf16.h>

typedef __attribute__((ext_vector_type(16))) __bf16 v16bf;
typedef __attribute__((ext_vector_type(8)))  float  v8f;

#define EPSF 1.1920929e-07f
#define BATCH 4096
#define DDIM  3072
#define NPAD  1024
#define NVAL  1023
#define CCLS  10

// ---------------- helpers ----------------
__device__ __forceinline__ unsigned short f2bfu(float f) {
    __bf16 h = (__bf16)f;
    return __builtin_bit_cast(unsigned short, h);
}
__device__ __forceinline__ float bfu2f(unsigned short u) {
    return (float)__builtin_bit_cast(__bf16, u);
}
__device__ __forceinline__ void split2(float f, unsigned short& h, unsigned short& l) {
    h = f2bfu(f);
    l = f2bfu(f - bfu2f(h));
}
__device__ __forceinline__ unsigned packhl(unsigned short a, unsigned short b) {
    return (unsigned)a | ((unsigned)b << 16);
}

union V16U { uint4 q[2]; v16bf v; };
__device__ __forceinline__ v16bf ld_frag(const unsigned short* p0, const unsigned short* p1) {
    V16U u;
    u.q[0] = *(const uint4*)p0;
    u.q[1] = *(const uint4*)p1;
    return u.v;
}

// ---------------- kernel 1: leaf softmax + zero accumulators ----------------
__global__ __launch_bounds__(256) void sdt_leaf_kernel(
    const float* __restrict__ leaf, float* __restrict__ Q, float* __restrict__ logQ,
    float* __restrict__ gNum, float* __restrict__ gDen, float* __restrict__ gLoss)
{
    int t = blockIdx.x * blockDim.x + threadIdx.x; // 0..1023
    if (t < 512) { gNum[t] = 0.f; gDen[t] = 0.f; }
    if (t == 0)  gLoss[0] = 0.f;
    if (t < 1024) {
        const float* r = leaf + t * CCLS;
        float m = r[0];
        #pragma unroll
        for (int c = 1; c < CCLS; ++c) m = fmaxf(m, r[c]);
        float e[CCLS]; float s = 0.f;
        #pragma unroll
        for (int c = 0; c < CCLS; ++c) { e[c] = expf(r[c] - m); s += e[c]; }
        float ls = logf(s);
        #pragma unroll
        for (int c = 0; c < CCLS; ++c) {
            Q[t * CCLS + c]    = e[c] / s;
            logQ[t * CCLS + c] = (r[c] - m) - ls;
        }
    }
}

// ---------------- kernel 2: split-bf16 WMMA GEMM + fused sigmoid ----------------
// probs[m][n] = sigmoid(beta[n] * (x[m,:] @ W[:,n] + b[n])), n < 1023; col 1023 = 0
__global__ __launch_bounds__(256) void sdt_gemm_kernel(
    const float* __restrict__ x, const float* __restrict__ W,
    const float* __restrict__ bias, const float* __restrict__ beta,
    float* __restrict__ probs)
{
    __shared__ __align__(16) unsigned short sAh[128][40]; // [row][k], 80B stride
    __shared__ __align__(16) unsigned short sAl[128][40];
    __shared__ __align__(16) unsigned short sBh[128][40]; // [col][k] (transposed W tile)
    __shared__ __align__(16) unsigned short sBl[128][40];

    const int tid   = threadIdx.x;
    const int lane  = tid & 31;
    const int wid   = tid >> 5;
    const int waveM = wid & 3;   // 4 waves over M (32 rows each)
    const int waveN = wid >> 2;  // 2 waves over N (64 cols each)
    const int n0 = blockIdx.x * 128;
    const int m0 = blockIdx.y * 128;

    v8f acc[2][4];
    #pragma unroll
    for (int i = 0; i < 2; ++i)
        #pragma unroll
        for (int j = 0; j < 4; ++j)
            acc[i][j] = v8f{0.f,0.f,0.f,0.f,0.f,0.f,0.f,0.f};

    const int arow  = tid >> 1;          // 0..127
    const int ahalf = (tid & 1) * 16;    // halfword offset 0 or 16
    const int bk2   = (tid & 15) * 2;    // k-pair base 0..30
    const int bcol  = (tid >> 4) * 8;    // col chunk 0..120

    for (int kb = 0; kb < DDIM; kb += 32) {
        __syncthreads();
        // ---- stage A (x tile, row-major, split hi/lo) ----
        {
            const float* xp = x + (size_t)(m0 + arow) * DDIM + kb + ahalf;
            if (kb + 32 < DDIM) __builtin_prefetch(xp + 32, 0, 0);
            unsigned hw[8], lw[8];
            #pragma unroll
            for (int i = 0; i < 4; ++i) {
                float4 v = ((const float4*)xp)[i];
                unsigned short h0,h1,h2,h3,l0,l1,l2,l3;
                split2(v.x,h0,l0); split2(v.y,h1,l1);
                split2(v.z,h2,l2); split2(v.w,h3,l3);
                hw[2*i]   = packhl(h0,h1); hw[2*i+1] = packhl(h2,h3);
                lw[2*i]   = packhl(l0,l1); lw[2*i+1] = packhl(l2,l3);
            }
            *(uint4*)&sAh[arow][ahalf]     = uint4{hw[0],hw[1],hw[2],hw[3]};
            *(uint4*)&sAh[arow][ahalf + 8] = uint4{hw[4],hw[5],hw[6],hw[7]};
            *(uint4*)&sAl[arow][ahalf]     = uint4{lw[0],lw[1],lw[2],lw[3]};
            *(uint4*)&sAl[arow][ahalf + 8] = uint4{lw[4],lw[5],lw[6],lw[7]};
        }
        // ---- stage B (W tile, transposed to [col][k], k-pairs packed in u32) ----
        {
            const float* w0 = W + (size_t)(kb + bk2) * NVAL + n0 + bcol;
            const float* w1 = w0 + NVAL;
            #pragma unroll
            for (int j = 0; j < 8; ++j) {
                int gc = n0 + bcol + j;
                float f0 = (gc < NVAL) ? w0[j] : 0.f;
                float f1 = (gc < NVAL) ? w1[j] : 0.f;
                unsigned short h0,l0,h1,l1;
                split2(f0,h0,l0); split2(f1,h1,l1);
                ((unsigned*)&sBh[bcol + j][0])[bk2 >> 1] = packhl(h0,h1);
                ((unsigned*)&sBl[bcol + j][0])[bk2 >> 1] = packhl(l0,l1);
            }
        }
        __syncthreads();

        // ---- fragments + WMMA ----
        const int la   = lane & 15;
        const int ksel = (lane >> 4) * 16;  // A halfword offset: lanes0-15 K0..15, lanes16-31 K16..31
        const int koff = (lane >> 4) * 16;  // B halfword offset: lanes0-15 K0..15, lanes16-31 K16..31

        v16bf bh[4], bl[4];
        #pragma unroll
        for (int nt = 0; nt < 4; ++nt) {
            int c = 64 * waveN + 16 * nt + la;
            bh[nt] = ld_frag(&sBh[c][koff], &sBh[c][koff + 8]);
            bl[nt] = ld_frag(&sBl[c][koff], &sBl[c][koff + 8]);
        }
        #pragma unroll
        for (int mt = 0; mt < 2; ++mt) {
            int r = 32 * waveM + 16 * mt + la;
            v16bf ah = ld_frag(&sAh[r][ksel], &sAh[r][ksel + 8]);
            v16bf al = ld_frag(&sAl[r][ksel], &sAl[r][ksel + 8]);
            #pragma unroll
            for (int nt = 0; nt < 4; ++nt) {
                acc[mt][nt] = __builtin_amdgcn_wmma_f32_16x16x32_bf16(
                    false, ah, false, bh[nt], (short)0, acc[mt][nt], false, false);
                acc[mt][nt] = __builtin_amdgcn_wmma_f32_16x16x32_bf16(
                    false, ah, false, bl[nt], (short)0, acc[mt][nt], false, false);
                acc[mt][nt] = __builtin_amdgcn_wmma_f32_16x16x32_bf16(
                    false, al, false, bh[nt], (short)0, acc[mt][nt], false, false);
            }
        }
    }

    // ---- epilogue: bias + beta + sigmoid, store probs ----
    const int nl = lane & 15;
    const int mh = (lane >> 4) * 8;
    #pragma unroll
    for (int nt = 0; nt < 4; ++nt) {
        int n = n0 + 64 * waveN + 16 * nt + nl;
        bool valid = (n < NVAL);
        float bia = valid ? bias[n] : 0.f;
        float bet = valid ? beta[n] : 0.f;
        #pragma unroll
        for (int mt = 0; mt < 2; ++mt) {
            int mbase = m0 + 32 * waveM + 16 * mt + mh;
            #pragma unroll
            for (int r = 0; r < 8; ++r) {
                float s = 0.f;
                if (valid) {
                    float z = bet * (acc[mt][nt][r] + bia);
                    s = 1.f / (1.f + expf(-z));
                }
                probs[(size_t)(mbase + r) * NPAD + n] = s;
            }
        }
    }
}

// ---------------- kernel 3: per-sample tree pass ----------------
__global__ __launch_bounds__(256) void sdt_tree_kernel(
    const float* __restrict__ probs, const float* __restrict__ y,
    const float* __restrict__ Q, const float* __restrict__ logQ,
    float* __restrict__ gNum, float* __restrict__ gDen, float* __restrict__ gLoss,
    float* __restrict__ out)
{
    __shared__ float sNum[512];
    __shared__ float sDen[512];
    __shared__ float sRed[256];
    const int t = threadIdx.x;
    for (int i = t; i < 512; i += 256) { sNum[i] = 0.f; sDen[i] = 0.f; }
    __syncthreads();

    const int b = blockIdx.x * 256 + t; // < 4096
    // label = argmax of one-hot row
    int label = 0; float mv = y[b * CCLS];
    #pragma unroll
    for (int c = 1; c < CCLS; ++c) {
        float v = y[b * CCLS + c];
        if (v > mv) { mv = v; label = c; }
    }

    const float* prow = probs + (size_t)b * NPAD;
    float pref[11]; pref[0] = 1.f;
    float loss = 0.f, best = -1.f; int bestLeaf = 0;

    for (int l = 0; l < 1024; ++l) {
        int dstart = 0;
        if (l) { int h = 31 - __clz(l ^ (l - 1)); dstart = 9 - h; }
        #pragma unroll
        for (int d = 0; d < 10; ++d) {
            if (d >= dstart) {
                int q   = l >> (10 - d);
                int idx = (1 << d) - 1 + q;
                float p = prow[idx];
                float P = pref[d];
                if (d < 9) { // nodes 0..510: alpha statistics (twice per sample/node)
                    atomicAdd(&sNum[idx], p * P);
                    atomicAdd(&sDen[idx], P);
                }
                int bit = (l >> (9 - d)) & 1;
                pref[d + 1] = P * (bit ? p : (1.f - p));
            }
        }
        float Pl = pref[10];
        loss += Pl * logQ[l * CCLS + label];
        if (Pl > best) { best = Pl; bestLeaf = l; }
    }

    // output row = Q[best_leaf]
    #pragma unroll
    for (int c = 0; c < CCLS; ++c)
        out[1 + (size_t)b * CCLS + c] = Q[bestLeaf * CCLS + c];

    // block-reduce loss
    sRed[t] = loss;
    __syncthreads();
    for (int s = 128; s > 0; s >>= 1) {
        if (t < s) sRed[t] += sRed[t + s];
        __syncthreads();
    }
    if (t == 0) atomicAdd(gLoss, sRed[0]);

    // flush alpha accumulators
    for (int i = t; i < 511; i += 256) {
        atomicAdd(&gNum[i], sNum[i]);
        atomicAdd(&gDen[i], sDen[i]);
    }
}

// ---------------- kernel 4: penalty + final scalar ----------------
__global__ __launch_bounds__(512) void sdt_final_kernel(
    const float* __restrict__ gNum, const float* __restrict__ gDen,
    const float* __restrict__ gLoss, float* __restrict__ out)
{
    __shared__ float red[512];
    const int t = threadIdx.x;
    float c = 0.f;
    if (t < 511) {
        int d = 31 - __clz(t + 1); // node depth 0..8
        float lm  = 0.1f * exp2f(-(float)(d + 1));
        float den = fmaxf(gDen[t], EPSF);
        float a   = gNum[t] / den;
        a = fminf(fmaxf(a, EPSF), 1.f - EPSF);
        c = -lm * 0.5f * (logf(a) + logf(1.f - a));
    }
    red[t] = c;
    __syncthreads();
    for (int s = 256; s > 0; s >>= 1) {
        if (t < s) red[t] += red[t + s];
        __syncthreads();
    }
    if (t == 0) out[0] = -(gLoss[0] / (float)BATCH) + red[0];
}

// ---------------- launcher ----------------
extern "C" void kernel_launch(void* const* d_in, const int* in_sizes, int n_in,
                              void* d_out, int out_size, void* d_ws, size_t ws_size,
                              hipStream_t stream) {
    const float* x    = (const float*)d_in[0]; // (4096, 3072)
    const float* y    = (const float*)d_in[1]; // (4096, 10)
    const float* W    = (const float*)d_in[2]; // (3072, 1023)
    const float* bia  = (const float*)d_in[3]; // (1023,)
    const float* beta = (const float*)d_in[4]; // (1023,)
    const float* leaf = (const float*)d_in[5]; // (1024, 10)
    float* out = (float*)d_out;                // [loss] + (4096,10)

    float* probs = (float*)d_ws;                       // 4096*1024
    float* Q     = probs + (size_t)BATCH * NPAD;       // 1024*10
    float* logQ  = Q + 1024 * CCLS;                    // 1024*10
    float* gNum  = logQ + 1024 * CCLS;                 // 512
    float* gDen  = gNum + 512;                         // 512
    float* gLoss = gDen + 512;                         // 1

    sdt_leaf_kernel<<<4, 256, 0, stream>>>(leaf, Q, logQ, gNum, gDen, gLoss);
    sdt_gemm_kernel<<<dim3(NPAD / 128, BATCH / 128), 256, 0, stream>>>(x, W, bia, beta, probs);
    sdt_tree_kernel<<<BATCH / 256, 256, 0, stream>>>(probs, y, Q, logQ, gNum, gDen, gLoss, out);
    sdt_final_kernel<<<1, 512, 0, stream>>>(gNum, gDen, gLoss, out);
}
